// StableAcousticModel_64398739636360
// MI455X (gfx1250) — matrix operand order repaired and checked by reading, not compiled, not run on hardware
//
#include <hip/hip_runtime.h>

// ---------------------------------------------------------------------------
// StableAcousticModel forward for MI455X (gfx1250), bf16 WMMA everywhere.
// B=16, T=512, F=2048, H=256, MEL=80.
// ---------------------------------------------------------------------------

typedef __bf16 bf16;
typedef __attribute__((ext_vector_type(16))) __bf16 v16bf;
typedef __attribute__((ext_vector_type(8)))  float  v8f;

union Frag { v16bf v; bf16 e[16]; uint4 q[2]; };

__device__ __forceinline__ v8f wmma_bf(const Frag a, const Frag b, v8f c) {
  return __builtin_amdgcn_wmma_f32_16x16x32_bf16(false, a.v, false, b.v,
                                                 (short)0, c, false, false);
}
__device__ __forceinline__ v8f zero8() { v8f z = {0.f,0.f,0.f,0.f,0.f,0.f,0.f,0.f}; return z; }
__device__ __forceinline__ float sigm(float x) { return 1.f / (1.f + __expf(-x)); }
__device__ __forceinline__ float siluf(float x) { return x / (1.f + __expf(-x)); }

// A fragment (16x32 bf16, MxK): lane holds row M=lane&15; two contiguous 8-elem
// K runs at k0 + (lane>>4)*8 and +16.  Requires k0+32 <= row length.
__device__ __forceinline__ Frag load_a(const bf16* At, int lda, int k0, int lane) {
  Frag r; int mr = lane & 15, hi = (lane >> 4) & 1;
  const bf16* p = At + (size_t)mr * lda + k0 + hi * 8;
  r.q[0] = *(const uint4*)p;
  r.q[1] = *(const uint4*)(p + 16);
  return r;
}

// Guarded A fragment: per-lane row index + validity, K clipped to kmax.
__device__ __forceinline__ Frag load_a_guard(const bf16* Ab, int lda, int frow,
                                             int valid, int k0, int kmax, int lane) {
  Frag r; int hi = (lane >> 4) & 1;
  if (valid && (k0 + 32) <= kmax) {
    const bf16* p = Ab + (size_t)frow * lda + k0 + hi * 8;
    r.q[0] = *(const uint4*)p;
    r.q[1] = *(const uint4*)(p + 16);
  } else {
#pragma unroll
    for (int j = 0; j < 16; ++j) {
      int k = k0 + hi * 8 + (j & 7) + ((j >= 8) ? 16 : 0);
      r.e[j] = (valid && k < kmax) ? Ab[(size_t)frow * lda + k] : (bf16)0.f;
    }
  }
  return r;
}

// A fragment from f32 source (row-major), converting to bf16; K clipped to kmax.
__device__ __forceinline__ Frag load_a_f32(const float* At, int lda, int k0,
                                           int kmax, int lane) {
  Frag r; int mr = lane & 15, hi = (lane >> 4) & 1;
  const float* p = At + (size_t)mr * lda + k0 + hi * 8;
#pragma unroll
  for (int c = 0; c < 2; ++c) {
    int kb = k0 + hi * 8 + c * 16;
    const float* pc = p + c * 16;
    if (kb + 8 <= kmax) {
      float4 f0 = *(const float4*)pc;
      float4 f1 = *(const float4*)(pc + 4);
      r.e[c*8+0]=(bf16)f0.x; r.e[c*8+1]=(bf16)f0.y; r.e[c*8+2]=(bf16)f0.z; r.e[c*8+3]=(bf16)f0.w;
      r.e[c*8+4]=(bf16)f1.x; r.e[c*8+5]=(bf16)f1.y; r.e[c*8+6]=(bf16)f1.z; r.e[c*8+7]=(bf16)f1.w;
    } else {
#pragma unroll
      for (int j = 0; j < 8; ++j) {
        int k = kb + j;
        r.e[c*8+j] = (k < kmax) ? (bf16)pc[j] : (bf16)0.f;
      }
    }
  }
  return r;
}

// B fragment (32x16 bf16, KxN): lane = column nt*16+(lane&15); two 8-elem K
// runs at kt*32 + (lane>>4)*8 and +16.  Pre-swizzled so this is one 32B load.
__device__ __forceinline__ Frag load_bfrag(const bf16* Bs, int nt, int kt, int KT, int lane) {
  const bf16* p = Bs + (((size_t)nt * KT + kt) * 32 + lane) * 16;
  Frag r;
  r.q[0] = ((const uint4*)p)[0];
  r.q[1] = ((const uint4*)p)[1];
  return r;
}

// ---------------------------------------------------------------------------
// Weight pre-swizzle: src element (k,n) = src[k*sk + n*sn]; zero-pad k >= K.
// dst layout: [nt][kt][lane][16] matching load_bfrag.
// ---------------------------------------------------------------------------
__global__ void swizzle_b(const float* __restrict__ src, bf16* __restrict__ dst,
                          int K, int KT, int sk, int sn, int tot) {
  int i = blockIdx.x * blockDim.x + threadIdx.x;
  if (i >= tot) return;
  int j = i & 15, lane = (i >> 4) & 31, frag = i >> 9;
  int kt = frag % KT, nt = frag / KT;
  int n = nt * 16 + (lane & 15);
  int k = kt * 32 + ((lane >> 4) & 1) * 8 + (j & 7) + ((j >= 8) ? 16 : 0);
  float v = (k < K) ? src[(size_t)k * sk + (size_t)n * sn] : 0.f;
  dst[i] = (bf16)v;
}

// ---------------------------------------------------------------------------
// Duration cumsum (inclusive), per batch row.
// ---------------------------------------------------------------------------
__global__ void cumsum_k(const int* __restrict__ dur, int* __restrict__ cum) {
  int b = blockIdx.x;
  if (threadIdx.x == 0) {
    int s = 0;
    for (int t = 0; t < 512; ++t) { s += dur[b * 512 + t]; cum[b * 512 + t] = s; }
  }
}

// ---------------------------------------------------------------------------
// Fused: x = bert@W_in + pros@W_pros + biases + emo; LN_in; LN_tok;
//        h = silu(.@W_tok1+b1); tok = silu(h@W_tok2+b2)  -> bf16 (B*T, 256)
// One block per 16-row tile; 8 waves x 2 n-tiles = 256 cols.
// ---------------------------------------------------------------------------
__global__ void input_proj_k(const float* __restrict__ bert, const float* __restrict__ pros,
                             const int* __restrict__ emo_ids, const float* __restrict__ emo_table,
                             const bf16* WsIn, const bf16* WsPros, const bf16* WsT1, const bf16* WsT2,
                             const float* b_in, const float* b_pros,
                             const float* g1, const float* bb1, const float* g2, const float* bb2,
                             const float* b_t1, const float* b_t2, bf16* __restrict__ tok) {
  __shared__ float xs[16][264];
  __shared__ bf16 hb[16][256];
  __shared__ float mu_s[16], rs_s[16];
  int tid = threadIdx.x, lane = tid & 31, wid = tid >> 5;
  int lcol = lane & 15, hi = (lane >> 4) & 1;
  int mt = blockIdx.x;
  int nt0 = wid * 2, nt1 = nt0 + 1;

  v8f acc0 = zero8(), acc1 = zero8();
  const float* At = bert + (size_t)mt * 16 * 768;
  for (int kt = 0; kt < 24; ++kt) {
    Frag a = load_a_f32(At, 768, kt * 32, 768, lane);
    acc0 = wmma_bf(a, load_bfrag(WsIn, nt0, kt, 24, lane), acc0);
    acc1 = wmma_bf(a, load_bfrag(WsIn, nt1, kt, 24, lane), acc1);
  }
  {
    Frag a = load_a_f32(pros + (size_t)mt * 16 * 20, 20, 0, 20, lane);
    acc0 = wmma_bf(a, load_bfrag(WsPros, nt0, 0, 1, lane), acc0);
    acc1 = wmma_bf(a, load_bfrag(WsPros, nt1, 0, 1, lane), acc1);
  }
  int bidx = (mt * 16) / 512;
  int eid = emo_ids[bidx];
#pragma unroll
  for (int r = 0; r < 8; ++r) {
    int mr = r + hi * 8;
    int n0 = nt0 * 16 + lcol, n1 = n0 + 16;
    xs[mr][n0] = acc0[r] + b_in[n0] + b_pros[n0] + emo_table[eid * 256 + n0];
    xs[mr][n1] = acc1[r] + b_in[n1] + b_pros[n1] + emo_table[eid * 256 + n1];
  }
  __syncthreads();
  if (tid < 16) {                       // LN_in stats
    float s = 0.f, s2 = 0.f;
    for (int n = 0; n < 256; ++n) { float v = xs[tid][n]; s += v; s2 += v * v; }
    float mu = s * (1.f / 256.f);
    mu_s[tid] = mu; rs_s[tid] = rsqrtf(s2 * (1.f / 256.f) - mu * mu + 1e-5f);
  }
  __syncthreads();
  for (int q = 0; q < 16; ++q) {        // apply LN_in
    int flat = q * 256 + tid; int mr = flat >> 8; int n = flat & 255;
    xs[mr][n] = (xs[mr][n] - mu_s[mr]) * rs_s[mr] * g1[n] + bb1[n];
  }
  __syncthreads();
  if (tid < 16) {                       // LN_tok stats
    float s = 0.f, s2 = 0.f;
    for (int n = 0; n < 256; ++n) { float v = xs[tid][n]; s += v; s2 += v * v; }
    float mu = s * (1.f / 256.f);
    mu_s[tid] = mu; rs_s[tid] = rsqrtf(s2 * (1.f / 256.f) - mu * mu + 1e-5f);
  }
  __syncthreads();
  for (int q = 0; q < 16; ++q) {        // apply LN_tok -> bf16
    int flat = q * 256 + tid; int mr = flat >> 8; int n = flat & 255;
    hb[mr][n] = (bf16)((xs[mr][n] - mu_s[mr]) * rs_s[mr] * g2[n] + bb2[n]);
  }
  __syncthreads();
  // GEMM with W_tok1 + silu (K=256)
  acc0 = zero8(); acc1 = zero8();
  for (int kt = 0; kt < 8; ++kt) {
    Frag a = load_a(&hb[0][0], 256, kt * 32, lane);
    acc0 = wmma_bf(a, load_bfrag(WsT1, nt0, kt, 8, lane), acc0);
    acc1 = wmma_bf(a, load_bfrag(WsT1, nt1, kt, 8, lane), acc1);
  }
  __syncthreads();
#pragma unroll
  for (int r = 0; r < 8; ++r) {
    int mr = r + hi * 8; int n0 = nt0 * 16 + lcol, n1 = n0 + 16;
    hb[mr][n0] = (bf16)siluf(acc0[r] + b_t1[n0]);
    hb[mr][n1] = (bf16)siluf(acc1[r] + b_t1[n1]);
  }
  __syncthreads();
  // GEMM with W_tok2 + silu -> tok
  acc0 = zero8(); acc1 = zero8();
  for (int kt = 0; kt < 8; ++kt) {
    Frag a = load_a(&hb[0][0], 256, kt * 32, lane);
    acc0 = wmma_bf(a, load_bfrag(WsT2, nt0, kt, 8, lane), acc0);
    acc1 = wmma_bf(a, load_bfrag(WsT2, nt1, kt, 8, lane), acc1);
  }
#pragma unroll
  for (int r = 0; r < 8; ++r) {
    int m = mt * 16 + r + hi * 8; int n0 = nt0 * 16 + lcol, n1 = n0 + 16;
    tok[(size_t)m * 256 + n0] = (bf16)siluf(acc0[r] + b_t2[n0]);
    tok[(size_t)m * 256 + n1] = (bf16)siluf(acc1[r] + b_t2[n1]);
  }
}

// ---------------------------------------------------------------------------
// Duration expansion: binary search on cum, gather token rows into time-major
// xe[(f*16 + b)*256 + :].  8 threads per output row (32 bf16 each).
// ---------------------------------------------------------------------------
__global__ void expand_k(const bf16* __restrict__ tok, const int* __restrict__ cum,
                         bf16* __restrict__ xe) {
  int tid = blockIdx.x * blockDim.x + threadIdx.x;   // B*F*8 total
  int chunk = tid & 7; int row = tid >> 3;           // row in [0, B*F)
  int b = row >> 11, f = row & 2047;
  const int* c = cum + b * 512;
  int lo = 0, hi = 512;
  while (lo < hi) { int mid = (lo + hi) >> 1; if (c[mid] <= f) lo = mid + 1; else hi = mid; }
  int idx = lo < 511 ? lo : 511;
  const uint4* s = (const uint4*)(tok + ((size_t)b * 512 + idx) * 256 + chunk * 32);
  uint4* d = (uint4*)(xe + ((size_t)f * 16 + b) * 256 + chunk * 32);
  d[0] = s[0]; d[1] = s[1]; d[2] = s[2]; d[3] = s[3];
}

// ---------------------------------------------------------------------------
// Generic GEMM: out(bf16) = act(A(bf16,MxK) @ Bs + bias).  8 waves x 1 n-tile.
// grid = (M/16, N/128)
// ---------------------------------------------------------------------------
__global__ void gemm_k(const bf16* __restrict__ A, int lda, const bf16* __restrict__ Bs,
                       int KT, const float* __restrict__ bias, int act,
                       bf16* __restrict__ out, int ldo) {
  int tid = threadIdx.x, lane = tid & 31, wid = tid >> 5;
  int lcol = lane & 15, hi = (lane >> 4) & 1;
  int ntB = blockIdx.y * 8 + wid;
  const bf16* At = A + (size_t)blockIdx.x * 16 * lda;
  v8f acc = zero8();
  for (int kt = 0; kt < KT; ++kt) {
    Frag a = load_a(At, lda, kt * 32, lane);
    acc = wmma_bf(a, load_bfrag(Bs, ntB, kt, KT, lane), acc);
  }
#pragma unroll
  for (int r = 0; r < 8; ++r) {
    int m = blockIdx.x * 16 + r + hi * 8;
    int n = ntB * 16 + lcol;
    float v = acc[r] + bias[n];
    if (act) v = siluf(v);
    out[(size_t)m * ldo + n] = (bf16)v;
  }
}

// ---------------------------------------------------------------------------
// Persistent bi-LSTM direction: grid=2 (fwd/bwd), 1024 threads (32 waves).
// Each wave owns 32 gate columns; Whh fragments live in VGPRs for all steps.
// Dynamic LDS: h(16x256 bf16) + c(16x256 f32) + gates(16x1024 f32) = 90112 B.
// ---------------------------------------------------------------------------
__global__ void __launch_bounds__(1024)
lstm_k(const bf16* __restrict__ preF, const bf16* __restrict__ preB,
       const bf16* __restrict__ WhhF, const bf16* __restrict__ WhhB,
       bf16* __restrict__ hout) {
  extern __shared__ char smem[];
  bf16*  h_s = (bf16*)smem;                      // 16*256
  float* c_s = (float*)(smem + 8192);            // 16*256
  float* g_s = (float*)(smem + 8192 + 16384);    // 16*1024

  int dir = blockIdx.x;
  const bf16* pre = dir ? preB : preF;
  const bf16* Whh = dir ? WhhB : WhhF;
  int colOff = dir * 256;

  int tid = threadIdx.x, lane = tid & 31, wid = tid >> 5;
  int lcol = lane & 15, hi = (lane >> 4) & 1;
  int nt0 = wid * 2, nt1 = nt0 + 1;

  Frag bf0[8], bf1[8];
#pragma unroll
  for (int kt = 0; kt < 8; ++kt) {
    bf0[kt] = load_bfrag(Whh, nt0, kt, 8, lane);
    bf1[kt] = load_bfrag(Whh, nt1, kt, 8, lane);
  }
#pragma unroll
  for (int q = 0; q < 4; ++q) { int flat = q * 1024 + tid; h_s[flat] = (bf16)0.f; c_s[flat] = 0.f; }
  __syncthreads();

  for (int s = 0; s < 2048; ++s) {
    int t = dir ? (2047 - s) : s;
    v8f a0 = zero8(), a1 = zero8();
#pragma unroll
    for (int kt = 0; kt < 8; ++kt) {
      Frag a = load_a(h_s, 256, kt * 32, lane);
      a0 = wmma_bf(a, bf0[kt], a0);
      a1 = wmma_bf(a, bf1[kt], a1);
    }
#pragma unroll
    for (int r = 0; r < 8; ++r) {
      int mr = r + hi * 8;
      g_s[mr * 1024 + nt0 * 16 + lcol] = a0[r];
      g_s[mr * 1024 + nt1 * 16 + lcol] = a1[r];
    }
    __syncthreads();
    const bf16* pt = pre + (size_t)t * 16384;    // 16*1024
#pragma unroll
    for (int q = 0; q < 4; ++q) {
      int flat = q * 1024 + tid; int mr = flat >> 8; int col = flat & 255;
      int gb = mr * 1024 + col;
      float iv = g_s[gb]       + (float)pt[gb];
      float fv = g_s[gb + 256] + (float)pt[gb + 256];
      float gv = g_s[gb + 512] + (float)pt[gb + 512];
      float ov = g_s[gb + 768] + (float)pt[gb + 768];
      float cn = sigm(fv) * c_s[flat] + sigm(iv) * tanhf(gv);
      float hn = sigm(ov) * tanhf(cn);
      c_s[flat] = cn;
      h_s[flat] = (bf16)hn;
      hout[((size_t)t * 16 + mr) * 512 + colOff + col] = (bf16)hn;
    }
    __syncthreads();
  }
}

// ---------------------------------------------------------------------------
// fh = LN_fr(h1 @ W_fp + b_fp)  -> bf16 (F*B, 256)
// ---------------------------------------------------------------------------
__global__ void fpln_k(const bf16* __restrict__ h1, const bf16* WsFp, const float* b_fp,
                       const float* g, const float* bb, bf16* __restrict__ fh) {
  __shared__ float xs[16][264];
  __shared__ float mu_s[16], rs_s[16];
  int tid = threadIdx.x, lane = tid & 31, wid = tid >> 5;
  int lcol = lane & 15, hi = (lane >> 4) & 1;
  int mt = blockIdx.x;
  int nt0 = wid * 2, nt1 = nt0 + 1;
  v8f acc0 = zero8(), acc1 = zero8();
  const bf16* At = h1 + (size_t)mt * 16 * 512;
  for (int kt = 0; kt < 16; ++kt) {
    Frag a = load_a(At, 512, kt * 32, lane);
    acc0 = wmma_bf(a, load_bfrag(WsFp, nt0, kt, 16, lane), acc0);
    acc1 = wmma_bf(a, load_bfrag(WsFp, nt1, kt, 16, lane), acc1);
  }
#pragma unroll
  for (int r = 0; r < 8; ++r) {
    int mr = r + hi * 8; int n0 = nt0 * 16 + lcol, n1 = n0 + 16;
    xs[mr][n0] = acc0[r] + b_fp[n0];
    xs[mr][n1] = acc1[r] + b_fp[n1];
  }
  __syncthreads();
  if (tid < 16) {
    float s = 0.f, s2 = 0.f;
    for (int n = 0; n < 256; ++n) { float v = xs[tid][n]; s += v; s2 += v * v; }
    float mu = s * (1.f / 256.f);
    mu_s[tid] = mu; rs_s[tid] = rsqrtf(s2 * (1.f / 256.f) - mu * mu + 1e-5f);
  }
  __syncthreads();
  for (int q = 0; q < 16; ++q) {
    int flat = q * 256 + tid; int mr = flat >> 8; int n = flat & 255;
    float v = (xs[mr][n] - mu_s[mr]) * rs_s[mr] * g[n] + bb[n];
    fh[((size_t)mt * 16 + mr) * 256 + n] = (bf16)v;
  }
}

// ---------------------------------------------------------------------------
// mel = fh@W_mel + b_mel + xe@W_skip + b_skip, written transposed to (B,F,80)
// as f32 (residual base) + bf16 (conv input).  5 active waves (N=80).
// ---------------------------------------------------------------------------
__global__ void mel_k(const bf16* __restrict__ fh, const bf16* __restrict__ xe,
                      const bf16* WsMel, const bf16* WsSkip,
                      const float* b_mel, const float* b_skip,
                      float* __restrict__ melb, bf16* __restrict__ m0) {
  int tid = threadIdx.x, lane = tid & 31, wid = tid >> 5;
  if (wid >= 5) return;
  int lcol = lane & 15, hi = (lane >> 4) & 1;
  int mt = blockIdx.x, nt = wid;
  v8f acc = zero8();
  const bf16* A1 = fh + (size_t)mt * 16 * 256;
  const bf16* A2 = xe + (size_t)mt * 16 * 256;
  for (int kt = 0; kt < 8; ++kt)
    acc = wmma_bf(load_a(A1, 256, kt * 32, lane), load_bfrag(WsMel, nt, kt, 8, lane), acc);
  for (int kt = 0; kt < 8; ++kt)
    acc = wmma_bf(load_a(A2, 256, kt * 32, lane), load_bfrag(WsSkip, nt, kt, 8, lane), acc);
#pragma unroll
  for (int r = 0; r < 8; ++r) {
    int m = mt * 16 + r + hi * 8;
    int f = m >> 4, b = m & 15;
    int n = nt * 16 + lcol;
    float v = acc[r] + b_mel[n] + b_skip[n];
    size_t o = ((size_t)b * 2048 + f) * 80 + n;
    melb[o] = v;
    m0[o] = (bf16)v;
  }
}

// ---------------------------------------------------------------------------
// Conv1d(k=5, SAME) as 5 shifted-GEMM taps.  A rows = frames of one batch,
// channels contiguous (lda = I).  Ws = 5 taps of swizzled (I, O) weights.
// ---------------------------------------------------------------------------
__global__ void conv_k(const bf16* __restrict__ A, int lda, int KT, int kmax,
                       const bf16* __restrict__ Ws, int NTtot,
                       const float* __restrict__ bias, int act,
                       bf16* __restrict__ outb, float* __restrict__ outf,
                       const float* __restrict__ resid) {
  int tid = threadIdx.x, lane = tid & 31, wid = tid >> 5;
  int ntB = blockIdx.y * 8 + wid;
  if (ntB >= NTtot) return;
  int lcol = lane & 15, hi = (lane >> 4) & 1;
  int mt = blockIdx.x;
  int m0r = mt * 16;
  int bb = m0r >> 11;          // batch
  int ft = m0r & 2047;         // frame tile base
  const bf16* Ab = A + (size_t)bb * 2048 * lda;
  v8f acc = zero8();
  size_t tapStride = (size_t)NTtot * KT * 512;
  for (int t = 0; t < 5; ++t) {
    int frow = ft + lcol + (t - 2);
    int valid = (frow >= 0) && (frow < 2048);
    const bf16* Wt = Ws + (size_t)t * tapStride;
    for (int kt = 0; kt < KT; ++kt) {
      Frag a = load_a_guard(Ab, lda, frow, valid, kt * 32, kmax, lane);
      acc = wmma_bf(a, load_bfrag(Wt, ntB, kt, KT, lane), acc);
    }
  }
#pragma unroll
  for (int r = 0; r < 8; ++r) {
    int m = m0r + r + hi * 8;
    int n = ntB * 16 + lcol;
    float v = acc[r] + bias[n];
    if (act) v = siluf(v);
    if (outb) outb[(size_t)m * (NTtot * 16) + n] = (bf16)v;
    if (outf) outf[(size_t)m * 80 + n] = resid[(size_t)m * 80 + n] + v;
  }
}

// ---------------------------------------------------------------------------
// Host orchestration
// ---------------------------------------------------------------------------
extern "C" void kernel_launch(void* const* d_in, const int* in_sizes, int n_in,
                              void* d_out, int out_size, void* d_ws, size_t ws_size,
                              hipStream_t stream) {
  (void)in_sizes; (void)n_in; (void)out_size; (void)ws_size;
  const float* bert  = (const float*)d_in[0];
  const float* pros  = (const float*)d_in[1];
  const int*   emoi  = (const int*)d_in[2];
  const int*   dur   = (const int*)d_in[3];
  const float* W_in  = (const float*)d_in[4];
  const float* b_in  = (const float*)d_in[5];
  const float* W_pr  = (const float*)d_in[6];
  const float* b_pr  = (const float*)d_in[7];
  const float* emot  = (const float*)d_in[8];
  const float* ln_in_g = (const float*)d_in[9];
  const float* ln_in_b = (const float*)d_in[10];
  const float* ln_tk_g = (const float*)d_in[11];
  const float* ln_tk_b = (const float*)d_in[12];
  const float* W_t1  = (const float*)d_in[13];
  const float* b_t1  = (const float*)d_in[14];
  const float* W_t2  = (const float*)d_in[15];
  const float* b_t2  = (const float*)d_in[16];
  const float* l0Wih = (const float*)d_in[17];
  const float* l0Whh = (const float*)d_in[18];
  const float* l0b   = (const float*)d_in[19];
  const float* l1Wih = (const float*)d_in[20];
  const float* l1Whh = (const float*)d_in[21];
  const float* l1b   = (const float*)d_in[22];
  const float* W_fp  = (const float*)d_in[23];
  const float* b_fp  = (const float*)d_in[24];
  const float* lnfr_g= (const float*)d_in[25];
  const float* lnfr_b= (const float*)d_in[26];
  const float* W_sk  = (const float*)d_in[27];
  const float* b_sk  = (const float*)d_in[28];
  const float* W_ml  = (const float*)d_in[29];
  const float* b_ml  = (const float*)d_in[30];
  const float* c1W   = (const float*)d_in[31];
  const float* c1b   = (const float*)d_in[32];
  const float* c2W   = (const float*)d_in[33];
  const float* c2b   = (const float*)d_in[34];
  const float* c3W   = (const float*)d_in[35];
  const float* c3b   = (const float*)d_in[36];

  char* base = (char*)d_ws; size_t off = 0;
  auto alloc = [&](size_t bytes) -> void* {
    void* p = base + off; off = (off + bytes + 255) & ~(size_t)255; return p;
  };
  // swizzled weights (bf16)
  bf16* wsIn  = (bf16*)alloc(196608 * 2);
  bf16* wsPr  = (bf16*)alloc(8192 * 2);
  bf16* wsT1  = (bf16*)alloc(65536 * 2);
  bf16* wsT2  = (bf16*)alloc(65536 * 2);
  bf16* wsIh0[2], *wsHh0[2], *wsIh1[2], *wsHh1[2];
  for (int d = 0; d < 2; ++d) {
    wsIh0[d] = (bf16*)alloc(262144 * 2);
    wsHh0[d] = (bf16*)alloc(262144 * 2);
    wsIh1[d] = (bf16*)alloc(524288 * 2);
    wsHh1[d] = (bf16*)alloc(262144 * 2);
  }
  bf16* wsFp = (bf16*)alloc(131072 * 2);
  bf16* wsSk = (bf16*)alloc(20480 * 2);
  bf16* wsMl = (bf16*)alloc(20480 * 2);
  bf16* wsC1 = (bf16*)alloc(5 * 24576 * 2);
  bf16* wsC2 = (bf16*)alloc(5 * 65536 * 2);
  bf16* wsC3 = (bf16*)alloc(5 * 20480 * 2);
  // activations
  int*  cum  = (int*)alloc(16 * 512 * 4);
  bf16* tok  = (bf16*)alloc((size_t)8192 * 256 * 2);
  bf16* xe   = (bf16*)alloc((size_t)32768 * 256 * 2 + 256);
  bf16* h0   = (bf16*)alloc((size_t)32768 * 512 * 2 + 256);
  bf16* h1   = (bf16*)alloc((size_t)32768 * 512 * 2 + 256);
  bf16* preA = (bf16*)alloc((size_t)32768 * 1024 * 2 + 256);
  bf16* preB = (bf16*)alloc((size_t)32768 * 1024 * 2 + 256);
  // reuse pre buffers after the LSTMs finish
  bf16*  fh   = (bf16*)preA;                                   // 16 MB
  float* melb = (float*)((char*)preA + 20u * 1024 * 1024);     // 10.5 MB
  bf16*  m0   = (bf16*) ((char*)preA + 34u * 1024 * 1024);     // 5.25 MB
  bf16*  y1   = (bf16*)preB;                                   // 16 MB
  bf16*  y2   = (bf16*) ((char*)preB + 20u * 1024 * 1024);     // 16 MB

  auto sw = [&](const float* src, bf16* dst, int K, int KT, int NT, int sk, int sn) {
    int tot = NT * KT * 512;
    swizzle_b<<<(tot + 255) / 256, 256, 0, stream>>>(src, dst, K, KT, sk, sn, tot);
  };
  sw(W_in, wsIn, 768, 24, 16, 256, 1);
  sw(W_pr, wsPr, 20, 1, 16, 256, 1);
  sw(W_t1, wsT1, 256, 8, 16, 256, 1);
  sw(W_t2, wsT2, 256, 8, 16, 256, 1);
  for (int d = 0; d < 2; ++d) {
    sw(l0Wih + (size_t)d * 1024 * 256, wsIh0[d], 256, 8, 64, 1, 256);
    sw(l0Whh + (size_t)d * 1024 * 256, wsHh0[d], 256, 8, 64, 1, 256);
    sw(l1Wih + (size_t)d * 1024 * 512, wsIh1[d], 512, 16, 64, 1, 512);
    sw(l1Whh + (size_t)d * 1024 * 256, wsHh1[d], 256, 8, 64, 1, 256);
  }
  sw(W_fp, wsFp, 512, 16, 16, 256, 1);
  sw(W_sk, wsSk, 256, 8, 5, 80, 1);
  sw(W_ml, wsMl, 256, 8, 5, 80, 1);
  for (int t = 0; t < 5; ++t) {
    sw(c1W + t, wsC1 + (size_t)t * 24576, 80, 3, 16, 5, 400);
    sw(c2W + t, wsC2 + (size_t)t * 65536, 256, 8, 16, 5, 1280);
    sw(c3W + t, wsC3 + (size_t)t * 20480, 256, 8, 5, 5, 1280);
  }

  cumsum_k<<<16, 32, 0, stream>>>(dur, cum);

  input_proj_k<<<512, 256, 0, stream>>>(bert, pros, emoi, emot, wsIn, wsPr, wsT1, wsT2,
                                        b_in, b_pr, ln_in_g, ln_in_b, ln_tk_g, ln_tk_b,
                                        b_t1, b_t2, tok);

  expand_k<<<1024, 256, 0, stream>>>(tok, cum, xe);

  // layer 0: pre-activations then recurrence
  gemm_k<<<dim3(2048, 8), 256, 0, stream>>>(xe, 256, wsIh0[0], 8, l0b, 0, preA, 1024);
  gemm_k<<<dim3(2048, 8), 256, 0, stream>>>(xe, 256, wsIh0[1], 8, l0b + 1024, 0, preB, 1024);
  lstm_k<<<2, 1024, 90112, stream>>>(preA, preB, wsHh0[0], wsHh0[1], h0);

  // layer 1
  gemm_k<<<dim3(2048, 8), 256, 0, stream>>>(h0, 512, wsIh1[0], 16, l1b, 0, preA, 1024);
  gemm_k<<<dim3(2048, 8), 256, 0, stream>>>(h0, 512, wsIh1[1], 16, l1b + 1024, 0, preB, 1024);
  lstm_k<<<2, 1024, 90112, stream>>>(preA, preB, wsHh1[0], wsHh1[1], h1);

  fpln_k<<<2048, 256, 0, stream>>>(h1, wsFp, b_fp, lnfr_g, lnfr_b, fh);

  mel_k<<<2048, 256, 0, stream>>>(fh, xe, wsMl, wsSk, b_ml, b_sk, melb, m0);

  conv_k<<<dim3(2048, 2), 256, 0, stream>>>(m0, 80, 3, 80, wsC1, 16, c1b, 1, y1, nullptr, nullptr);
  conv_k<<<dim3(2048, 2), 256, 0, stream>>>(y1, 256, 8, 256, wsC2, 16, c2b, 1, y2, nullptr, nullptr);
  conv_k<<<dim3(2048, 1), 256, 0, stream>>>(y2, 256, 8, 256, wsC3, 5, c3b, 0, nullptr,
                                            (float*)d_out, melb);
}
